// Transformer_13451837571753
// MI455X (gfx1250) — matrix operand order repaired and checked by reading, not compiled, not run on hardware
//
#include <hip/hip_runtime.h>
#include <hip/hip_bf16.h>
#include <cmath>
#include <cstdint>

// ---------------------------------------------------------------------------
// MI455X (gfx1250) transformer forward, v2.
//
// Compute-bound (~275 GFLOP vs <0.5 GB HBM): all GEMMs on
// v_wmma_f32_16x16x32_bf16. v2: operands pre-converted to bf16 (weights
// transposed to [N,K] once), so LDS staging is pure 16B-chunk copies done with
// global_load_async_to_lds_b128 (ASYNCcnt) into double-buffered LDS; zero
// conversion VALU in the GEMM hot loop.
// ---------------------------------------------------------------------------

#define NUM_LAYERS 4
#define D_MODEL    512
#define NUM_HEADS  8
#define DEPTH      64
#define DFF        2048
#define BB         4
#define SS         1024
#define BS         (BB * SS)            // 4096 rows

typedef __attribute__((ext_vector_type(16))) __bf16 bfrag_t;
typedef __attribute__((ext_vector_type(8)))  float  v8f;
typedef unsigned short ushort_t;

__device__ __forceinline__ unsigned short f2bf(float f) {   // fp32 -> bf16 RNE
    unsigned int u = __float_as_uint(f);
    u += 0x7FFFu + ((u >> 16) & 1u);
    return (unsigned short)(u >> 16);
}
__device__ __forceinline__ float bf2f(unsigned short u) {
    return __uint_as_float((unsigned int)u << 16);
}

__device__ __forceinline__ void wait_async0() {
#if __has_builtin(__builtin_amdgcn_s_wait_asynccnt)
    __builtin_amdgcn_s_wait_asynccnt(0);
#else
    asm volatile("s_wait_asynccnt 0x0" ::: "memory");
#endif
}

// one 16-byte async global->LDS chunk (per-lane addresses, ASYNCcnt tracked)
__device__ __forceinline__ void async_chunk16(const ushort_t* g, ushort_t* lds_generic) {
    unsigned loff = (unsigned)(uintptr_t)lds_generic;   // low 32 bits = LDS offset
    asm volatile("global_load_async_to_lds_b128 %0, %1, off"
                 :: "v"(loff), "v"(g) : "memory");
}

// ------------------------------ GEMM ---------------------------------------
// C = alpha * (A @ B^T + bias), A:[M,K] bf16 row-major, B:[N,K] bf16 row-major.

struct GemmP {
    const ushort_t* A;
    const ushort_t* B;
    float*    Cf;               // fp32 out (nullable)
    ushort_t* Cb;               // bf16 row-major out (nullable)
    ushort_t* Ct;               // bf16 [b,h,d,s] transposed out (nullable)
    const float* bias;          // nullable
    int M, N, K;
    long long lda, ldb, ldc;
    int zInner;
    long long sAo, sAi, sBo, sBi, sCo, sCi;
    float alpha;
    int relu;
};

#define LDSW 40   // bf16 elems per LDS row (32 data + 8 pad) = 80B, 16B aligned

// issue the 2 async 16B chunks this thread owns for a 128x32 tile
__device__ __forceinline__ void stage_tile(const ushort_t* G, long long ld,
                                           int r0, int k0, ushort_t* lds,
                                           int tid, int limit, bool bounds) {
    #pragma unroll
    for (int i = 0; i < 2; i++) {
        const int linear = tid + (i << 8);
        const int r = linear >> 2;          // 0..127
        const int c = (linear & 3) << 3;    // 0,8,16,24
        if (!bounds || (r0 + r) < limit) {
            async_chunk16(G + (long long)(r0 + r) * ld + (k0 + c),
                          lds + (r * LDSW + c));
        }
    }
}

// Block tile 128x128x32, 256 threads = 8 waves (2M x 4N), 8 WMMA per wave.
template<bool BOUNDS>
__global__ __launch_bounds__(256) void gemm_wmma(GemmP p) {
    __shared__ __align__(16) ushort_t sA[2][128 * LDSW];
    __shared__ __align__(16) ushort_t sB[2][128 * LDSW];

    const int tid  = threadIdx.x;
    const int lane = tid & 31;
    const int wave = tid >> 5;
    const int wm   = wave >> 2;
    const int wn   = wave & 3;

    const int zo = (int)blockIdx.z / p.zInner;
    const int zi = (int)blockIdx.z % p.zInner;
    const ushort_t* A = p.A + (long long)zo * p.sAo + (long long)zi * p.sAi;
    const ushort_t* B = p.B + (long long)zo * p.sBo + (long long)zi * p.sBi;

    const int m0 = blockIdx.y * 128;
    const int n0 = blockIdx.x * 128;

    v8f acc[4][2] = {};

    const int TK = p.K >> 5;                 // K is always a multiple of 32

    // prologue: stage tile 0 into buffer 0
    stage_tile(A, p.lda, m0, 0, sA[0], tid, p.M, false);
    stage_tile(B, p.ldb, n0, 0, sB[0], tid, p.N, BOUNDS);
    wait_async0();
    __syncthreads();

    const int rA = lane & 15;
    const int kA = (lane < 16) ? 0 : 8;      // A frag: chunks [kA,kA+8),[kA+16,kA+24)
    const int kB = (lane < 16) ? 0 : 8;      // B frag: chunks [kB,kB+8),[kB+16,kB+24)

    for (int kt = 0; kt < TK; kt++) {
        const int cur = kt & 1;
        if (kt + 1 < TK) {                   // prefetch next tile into other buffer
            const int k0 = (kt + 1) << 5;
            stage_tile(A, p.lda, m0, k0, sA[1 - cur], tid, p.M, false);
            stage_tile(B, p.ldb, n0, k0, sB[1 - cur], tid, p.N, BOUNDS);
        }

        union FragU { uint4 q[2]; bfrag_t v; };
        FragU a[4], b[2];
        #pragma unroll
        for (int i = 0; i < 4; i++) {
            const ushort_t* pa = &sA[cur][(wm * 64 + i * 16 + rA) * LDSW + kA];
            a[i].q[0] = *(const uint4*)pa;
            a[i].q[1] = *(const uint4*)(pa + 16);
        }
        #pragma unroll
        for (int j = 0; j < 2; j++) {
            const ushort_t* pb = &sB[cur][(wn * 32 + j * 16 + rA) * LDSW + kB];
            b[j].q[0] = *(const uint4*)pb;
            b[j].q[1] = *(const uint4*)(pb + 16);
        }
        #pragma unroll
        for (int i = 0; i < 4; i++)
            #pragma unroll
            for (int j = 0; j < 2; j++)
                acc[i][j] = __builtin_amdgcn_wmma_f32_16x16x32_bf16(
                    false, a[i].v, false, b[j].v, (short)0, acc[i][j], false, false);

        wait_async0();                       // next buffer fully landed
        __syncthreads();
    }

    // epilogue: alpha * (acc + bias), optional relu; fp32 / bf16 / bf16-T out
    float*    Cf = p.Cf ? p.Cf + (long long)zo * p.sCo + (long long)zi * p.sCi : nullptr;
    ushort_t* Cb = p.Cb ? p.Cb + (long long)zo * p.sCo + (long long)zi * p.sCi : nullptr;
    #pragma unroll
    for (int i = 0; i < 4; i++) {
        const int rbase = m0 + wm * 64 + i * 16 + ((lane >= 16) ? 8 : 0);
        #pragma unroll
        for (int j = 0; j < 2; j++) {
            const int col = n0 + wn * 32 + j * 16 + (lane & 15);
            if (BOUNDS && col >= p.N) continue;
            const float bv = p.bias ? p.bias[col] : 0.f;
            #pragma unroll
            for (int e = 0; e < 8; e++) {
                const int row = rbase + e;
                if (BOUNDS && row >= p.M) continue;
                float val = (acc[i][j][e] + bv) * p.alpha;
                if (p.relu) val = fmaxf(val, 0.f);
                const long long o = (long long)row * p.ldc + col;
                if (Cf) Cf[o] = val;
                if (Cb) Cb[o] = f2bf(val);
                if (p.Ct)                    // [b,h,d,s]: b=row>>10, h=col>>6
                    p.Ct[((long long)(row >> 10) * NUM_HEADS + (col >> 6)) *
                             (DEPTH * SS) +
                         (long long)(col & (DEPTH - 1)) * SS + (row & (SS - 1))] =
                        f2bf(val);
            }
        }
    }
}

// --------------------------- small kernels ---------------------------------

__global__ __launch_bounds__(256) void convert_bf16(
    ushort_t* __restrict__ dst, const float* __restrict__ src, int n) {
    const int g = blockIdx.x * 256 + threadIdx.x;
    if (g < n) dst[g] = f2bf(src[g]);
}

// dst[n*K + k] = bf16(src[k*N + n])   (weight [K,N] -> [N,K] bf16)
__global__ __launch_bounds__(256) void convert_transpose(
    ushort_t* __restrict__ dst, const float* __restrict__ src, int K, int N) {
    const int g = blockIdx.x * 256 + threadIdx.x;
    if (g < K * N) {
        const int k = g / N, n = g - k * N;
        dst[(long long)n * K + k] = f2bf(src[g]);
    }
}

// out = LayerNorm(x + res), 512 cols; optional bf16 mirror.
__global__ __launch_bounds__(256) void layernorm_res(
    float* __restrict__ out, ushort_t* __restrict__ outb,
    const float* __restrict__ x, const float* __restrict__ res,
    const float* __restrict__ g, const float* __restrict__ bta) {
    const int row = blockIdx.x, t = threadIdx.x;
    const size_t base = (size_t)row * D_MODEL;
    float v0 = x[base + t]       + res[base + t];
    float v1 = x[base + t + 256] + res[base + t + 256];
    __shared__ float s1[256], s2[256];
    s1[t] = v0 + v1;
    s2[t] = v0 * v0 + v1 * v1;
    __syncthreads();
    for (int st = 128; st > 0; st >>= 1) {
        if (t < st) { s1[t] += s1[t + st]; s2[t] += s2[t + st]; }
        __syncthreads();
    }
    const float mean = s1[0] * (1.0f / D_MODEL);
    const float var  = s2[0] * (1.0f / D_MODEL) - mean * mean;
    const float rstd = rsqrtf(var + 1e-6f);
    const float o0 = (v0 - mean) * rstd * g[t]       + bta[t];
    const float o1 = (v1 - mean) * rstd * g[t + 256] + bta[t + 256];
    out[base + t] = o0;  out[base + t + 256] = o1;
    if (outb) { outb[base + t] = f2bf(o0); outb[base + t + 256] = f2bf(o1); }
}

// softmax over q of bf16 logits[bh,k,q], +mask[b,q]*(-1e9);
// writes bf16 TRANSPOSED output outT[bh,q,k].
__global__ __launch_bounds__(256) void softmax_maskT(
    const ushort_t* __restrict__ logits, ushort_t* __restrict__ outT,
    const float* __restrict__ mask) {
    const int row = blockIdx.x, t = threadIdx.x;     // row = bh*1024 + k
    const int bh = row >> 10, k = row & (SS - 1);
    const ushort_t* p = logits + (size_t)row * SS;
    const float* m = mask + (size_t)(row >> 13) * SS;
    float v[4];
    float mx = -3.402823e38f;
    #pragma unroll
    for (int i = 0; i < 4; i++) {
        const int q = t + i * 256;
        v[i] = bf2f(p[q]) + m[q] * (-1e9f);
        mx = fmaxf(mx, v[i]);
    }
    __shared__ float red[256];
    red[t] = mx; __syncthreads();
    for (int st = 128; st > 0; st >>= 1) {
        if (t < st) red[t] = fmaxf(red[t], red[t + st]);
        __syncthreads();
    }
    mx = red[0]; __syncthreads();
    float s = 0.f;
    #pragma unroll
    for (int i = 0; i < 4; i++) { v[i] = expf(v[i] - mx); s += v[i]; }
    red[t] = s; __syncthreads();
    for (int st = 128; st > 0; st >>= 1) {
        if (t < st) red[t] += red[t + st];
        __syncthreads();
    }
    const float inv = 1.0f / red[0];
    #pragma unroll
    for (int i = 0; i < 4; i++) {
        const int q = t + i * 256;
        outT[((size_t)bh * SS + q) * SS + k] = f2bf(v[i] * inv);
    }
}

__global__ __launch_bounds__(256) void pool_sum(
    float* __restrict__ out, const float* __restrict__ x) {
    const int gidx = blockIdx.x * 256 + threadIdx.x;
    if (gidx < BB * D_MODEL) {
        const int b = gidx >> 9, d = gidx & (D_MODEL - 1);
        const float* p = x + (size_t)b * SS * D_MODEL + d;
        float s = 0.f;
        for (int i = 0; i < SS; i++) s += p[(size_t)i * D_MODEL];
        out[gidx] = s;
    }
}

__global__ __launch_bounds__(256) void final_head(
    float* __restrict__ out, const float* __restrict__ pooled,
    const float* __restrict__ w, const float* __restrict__ bias) {
    __shared__ float red[256];
    __shared__ float res[BB * 2];
    const int t = threadIdx.x;
    for (int o = 0; o < BB * 2; o++) {
        const int b = o >> 1, c = o & 1;
        float s = 0.f;
        for (int k = t; k < D_MODEL; k += 256)
            s += pooled[b * D_MODEL + k] * w[k * 2 + c];
        red[t] = s; __syncthreads();
        for (int st = 128; st > 0; st >>= 1) {
            if (t < st) red[t] += red[t + st];
            __syncthreads();
        }
        if (t == 0) res[o] = red[0] + bias[c];
        __syncthreads();
    }
    if (t == 0) {
        float ss = 0.f;
        for (int o = 0; o < BB * 2; o++) ss += res[o] * res[o];
        const float nrm = sqrtf(fmaxf(ss, 1e-12f));
        for (int o = 0; o < BB * 2; o++) out[o] = res[o] / nrm;
    }
}

// ------------------------------ host side ----------------------------------

struct DenseP { const float* w; const float* b; };
struct MhaP   { DenseP wq, wk, wv, out; };
struct LnPar  { const float* g; const float* b; };
struct LayerP { MhaP mha; DenseP ffn1, ffn2; LnPar lnA, lnB; };
struct LayerW { ushort_t *wq, *wk, *wv, *wo, *w1, *w2; };

static void launch_gemm(hipStream_t s, bool bounds,
                        const ushort_t* A, const ushort_t* B,
                        float* Cf, ushort_t* Cb, ushort_t* Ct, const float* bias,
                        int M, int N, int K,
                        long long lda, long long ldb, long long ldc,
                        int batch, int zInner,
                        long long sAo, long long sAi, long long sBo, long long sBi,
                        long long sCo, long long sCi,
                        float alpha, int relu) {
    GemmP p{A, B, Cf, Cb, Ct, bias, M, N, K, lda, ldb, ldc,
            zInner > 0 ? zInner : 1, sAo, sAi, sBo, sBi, sCo, sCi, alpha, relu};
    dim3 g((N + 127) / 128, (M + 127) / 128, batch);
    if (bounds) gemm_wmma<true ><<<g, dim3(256), 0, s>>>(p);
    else        gemm_wmma<false><<<g, dim3(256), 0, s>>>(p);
}

extern "C" void kernel_launch(void* const* d_in, const int* in_sizes, int n_in,
                              void* d_out, int out_size, void* d_ws, size_t ws_size,
                              hipStream_t stream) {
    (void)in_sizes; (void)n_in; (void)out_size; (void)ws_size;

    const float* inp         = (const float*)d_in[0];
    const float* tar         = (const float*)d_in[1];
    const float* enc_mask    = (const float*)d_in[2];
    const float* cross_small = (const float*)d_in[3];
    const float* cross_mask  = (const float*)d_in[4];

    int idx = 5;
    auto nf = [&]() { return (const float*)d_in[idx++]; };
    DenseP enc_first{nf(), nf()};
    DenseP dec_first{nf(), nf()};
    LayerP enc[NUM_LAYERS], dec[NUM_LAYERS];
    for (int l = 0; l < NUM_LAYERS; l++) {
        enc[l].mha  = MhaP{{nf(), nf()}, {nf(), nf()}, {nf(), nf()}, {nf(), nf()}};
        enc[l].ffn1 = DenseP{nf(), nf()};
        enc[l].ffn2 = DenseP{nf(), nf()};
        enc[l].lnA  = LnPar{nf(), nf()};
        enc[l].lnB  = LnPar{nf(), nf()};
    }
    for (int l = 0; l < NUM_LAYERS; l++) {
        dec[l].mha  = MhaP{{nf(), nf()}, {nf(), nf()}, {nf(), nf()}, {nf(), nf()}};
        dec[l].ffn1 = DenseP{nf(), nf()};
        dec[l].ffn2 = DenseP{nf(), nf()};
        dec[l].lnA  = LnPar{nf(), nf()};
        dec[l].lnB  = LnPar{nf(), nf()};
    }
    DenseP fin{nf(), nf()};

    // ---------------- workspace: fp32 region then bf16 region ----------------
    float* Wf = (float*)d_ws;
    size_t fo = 0;
    auto allocF = [&](size_t n) { float* p = Wf + fo; fo += n; return p; };
    const size_t XS = (size_t)BS * D_MODEL;                 // 2M elems
    float* X    = allocF(XS);       // encoder stream / enc_out (fp32)
    float* XD   = allocF(XS);       // decoder stream (fp32)
    float* PROJ = allocF(XS);
    float* OUT1 = allocF(XS);
    float* FF   = allocF(XS);
    float* POOL = allocF((size_t)BB * D_MODEL);

    ushort_t* Wh = (ushort_t*)(Wf + ((fo + 7) & ~(size_t)7));
    size_t ho = 0;
    auto allocH = [&](size_t n) { ushort_t* p = Wh + ho; ho += (n + 7) & ~(size_t)7; return p; };
    ushort_t* Xb    = allocH(XS);
    ushort_t* XDb   = allocH(XS);
    ushort_t* Qb    = allocH(XS);
    ushort_t* Kb    = allocH(XS);
    ushort_t* VT    = allocH(XS);                           // [b,h,d,s]
    ushort_t* ATTb  = allocH(XS);
    ushort_t* OUT1b = allocH(XS);
    ushort_t* PROJb = allocH(XS);
    ushort_t* HIDb  = allocH((size_t)BS * DFF);
    ushort_t* LOG   = allocH((size_t)BB * NUM_HEADS * SS * SS);   // [bh,k,q]
    ushort_t* LOGT  = allocH((size_t)BB * NUM_HEADS * SS * SS);   // [bh,q,k]
    ushort_t* inpb  = allocH((size_t)BS * 64);
    ushort_t* tarb  = allocH((size_t)BS * 64);

    auto prepW = [&](DenseP d, int K, int N) {              // [K,N] fp32 -> [N,K] bf16
        ushort_t* t = allocH((size_t)K * N);
        const int tot = K * N;
        convert_transpose<<<(tot + 255) / 256, 256, 0, stream>>>(t, d.w, K, N);
        return t;
    };
    convert_bf16<<<(BS * 64 + 255) / 256, 256, 0, stream>>>(inpb, inp, BS * 64);
    convert_bf16<<<(BS * 64 + 255) / 256, 256, 0, stream>>>(tarb, tar, BS * 64);
    ushort_t* encFirstT = prepW(enc_first, 64, D_MODEL);
    ushort_t* decFirstT = prepW(dec_first, 64, D_MODEL);
    LayerW encW[NUM_LAYERS], decW[NUM_LAYERS];
    for (int l = 0; l < NUM_LAYERS; l++) {
        encW[l].wq = prepW(enc[l].mha.wq, D_MODEL, D_MODEL);
        encW[l].wk = prepW(enc[l].mha.wk, D_MODEL, D_MODEL);
        encW[l].wv = prepW(enc[l].mha.wv, D_MODEL, D_MODEL);
        encW[l].wo = prepW(enc[l].mha.out, D_MODEL, D_MODEL);
        encW[l].w1 = prepW(enc[l].ffn1, D_MODEL, DFF);
        encW[l].w2 = prepW(enc[l].ffn2, DFF, D_MODEL);
    }
    for (int l = 0; l < NUM_LAYERS; l++) {
        decW[l].wq = prepW(dec[l].mha.wq, D_MODEL, D_MODEL);
        decW[l].wk = prepW(dec[l].mha.wk, D_MODEL, D_MODEL);
        decW[l].wv = prepW(dec[l].mha.wv, D_MODEL, D_MODEL);
        decW[l].wo = prepW(dec[l].mha.out, D_MODEL, D_MODEL);
        decW[l].w1 = prepW(dec[l].ffn1, D_MODEL, DFF);
        decW[l].w2 = prepW(dec[l].ffn2, DFF, D_MODEL);
    }

    const float inv_sqrt_dm = 1.0f / sqrtf((float)D_MODEL);
    const long long SD  = (long long)SS * D_MODEL;
    const long long SSQ = (long long)SS * SS;

    // dense: A bf16 [M,K], W^T bf16 [N,K]
    auto dense = [&](const ushort_t* a, const ushort_t* wT, const float* bias,
                     float* cf, ushort_t* cb, ushort_t* ct,
                     int M, int K, int N, float alpha, int relu) {
        launch_gemm(stream, false, a, wT, cf, cb, ct, bias, M, N, K, K, K, N,
                    1, 1, 0, 0, 0, 0, 0, 0, alpha, relu);
    };

    // mha: V,K from kv_b; Q from q_b; result PROJ (fp32) + PROJb (bf16)
    auto attention = [&](const ushort_t* kv_b, const ushort_t* q_b,
                         const MhaP& m, const LayerW& w, const float* mask) {
        dense(kv_b, w.wv, m.wv.b, nullptr, nullptr, VT, BS, D_MODEL, D_MODEL, 1.f, 0);
        dense(kv_b, w.wk, m.wk.b, nullptr, Kb, nullptr, BS, D_MODEL, D_MODEL, 1.f, 0);
        dense(q_b,  w.wq, m.wq.b, nullptr, Qb, nullptr, BS, D_MODEL, D_MODEL, 1.f, 0);
        // logits[bh,k,q] = K @ Q^T / 8  (batched over b,h; contiguous depth-K)
        launch_gemm(stream, false, Kb, Qb, nullptr, LOG, nullptr, nullptr,
                    SS, SS, DEPTH, D_MODEL, D_MODEL, SS,
                    BB * NUM_HEADS, NUM_HEADS, SD, DEPTH, SD, DEPTH,
                    (long long)NUM_HEADS * SSQ, SSQ, 0.125f, 0);
        softmax_maskT<<<BB * NUM_HEADS * SS, 256, 0, stream>>>(LOG, LOGT, mask);
        // out[b,q,h*64+d] = attnT[q,:] . VT[d,:]  (contiguous s-K on both sides)
        launch_gemm(stream, true, LOGT, VT, nullptr, ATTb, nullptr, nullptr,
                    SS, DEPTH, SS, SS, SS, D_MODEL,
                    BB * NUM_HEADS, NUM_HEADS,
                    (long long)NUM_HEADS * SSQ, SSQ,
                    (long long)NUM_HEADS * DEPTH * SS, (long long)DEPTH * SS,
                    SD, DEPTH, 1.f, 0);
        dense(ATTb, w.wo, m.out.b, PROJ, PROJb, nullptr, BS, D_MODEL, D_MODEL, 1.f, 0);
    };

    // ------------------------------ encoder ------------------------------
    dense(inpb, encFirstT, enc_first.b, X, Xb, nullptr, BS, 64, D_MODEL, inv_sqrt_dm, 0);
    for (int l = 0; l < NUM_LAYERS; l++) {
        const LayerP& L = enc[l];  const LayerW& W = encW[l];
        attention(Xb, Xb, L.mha, W, enc_mask);
        layernorm_res<<<BS, 256, 0, stream>>>(OUT1, OUT1b, X, PROJ, L.lnA.g, L.lnA.b);
        dense(OUT1b, W.w1, L.ffn1.b, nullptr, HIDb, nullptr, BS, D_MODEL, DFF, 1.f, 1);
        dense(HIDb,  W.w2, L.ffn2.b, FF, nullptr, nullptr, BS, DFF, D_MODEL, 1.f, 0);
        layernorm_res<<<BS, 256, 0, stream>>>(X, Xb, OUT1, FF, L.lnB.g, L.lnB.b);
    }
    // X/Xb now hold enc_out

    // ------------------------------ decoder ------------------------------
    dense(tarb, decFirstT, dec_first.b, XD, XDb, nullptr, BS, 64, D_MODEL, inv_sqrt_dm, 0);
    for (int l = 0; l < NUM_LAYERS; l++) {
        const LayerP& L = dec[l];  const LayerW& W = decW[l];
        const float* mk = (l == 0) ? cross_mask : cross_small;
        attention(XDb, Xb, L.mha, W, mk);          // v_in=k_in=xd, q_in=enc_out
        const float* out2f; const ushort_t* out2b;
        if (l == 0) { out2f = PROJ; out2b = PROJb; }   // no residual/LN on block 0
        else {
            layernorm_res<<<BS, 256, 0, stream>>>(OUT1, OUT1b, PROJ, XD, L.lnA.g, L.lnA.b);
            out2f = OUT1; out2b = OUT1b;
        }
        dense(out2b, W.w1, L.ffn1.b, nullptr, HIDb, nullptr, BS, D_MODEL, DFF, 1.f, 1);
        dense(HIDb,  W.w2, L.ffn2.b, FF, nullptr, nullptr, BS, DFF, D_MODEL, 1.f, 0);
        layernorm_res<<<BS, 256, 0, stream>>>(XD, XDb, FF, out2f, L.lnB.g, L.lnB.b);
    }

    // ----------------------- pool + head + l2 norm -----------------------
    pool_sum<<<(BB * D_MODEL + 255) / 256, 256, 0, stream>>>(POOL, XD);
    final_head<<<1, 256, 0, stream>>>((float*)d_out, POOL, fin.w, fin.b);
}
